// ForwardModel_36275293782592
// MI455X (gfx1250) — matrix operand order repaired and checked by reading, not compiled, not run on hardware
//
#include <hip/hip_runtime.h>
#include <math.h>

typedef __attribute__((ext_vector_type(16))) _Float16     v16h;
typedef __attribute__((ext_vector_type(8)))  float        v8f;
typedef __attribute__((ext_vector_type(4)))  unsigned int u32x4;

#define GAS __attribute__((address_space(1)))
typedef const GAS u32x4*    g_cu4;   // global-AS 16B load pointer
typedef       GAS _Float16* g_h;     // global-AS f16 store pointer

#if defined(__has_builtin)
#  if __has_builtin(__builtin_amdgcn_sched_barrier)
#    define SCHED_FENCE() __builtin_amdgcn_sched_barrier(0)
#  endif
#endif
#ifndef SCHED_FENCE
#  define SCHED_FENCE()
#endif

#define NS        2048      // hidden states
#define NOBS      8192      // emission alphabet
#define TT        8192      // sequence length
#define NBLK      128       // persistent blocks, one 16-column tile each
#define NTHREADS  512       // 16 wave32s per tile
#define NWAVE_TILE 16
#define KC_TOTAL  64        // 2048 / 32 K-chunks
#define KC_PER_WAVE 4       // short dependent chain per wave
#define ALPHA_SCALE 1024.0f // keeps ~5e-4 probabilities in f16 normal range
#define ALPHA_INVS  (1.0f / 1024.0f)

// ---------------------------------------------------------------------------
// Prologue: repack fp32 A[k][i] (row-major 2048x2048) into f16 in the exact
// per-lane WMMA B-operand order.  Linear index = ((ct*64+kc)*32+lane)*16+h,
//   k = kc*32 + (lane>=16 ? 16 : 0) + h ,  i = ct*16 + (lane & 15)
// ---------------------------------------------------------------------------
__global__ void pack_A16(const float* __restrict__ A, _Float16* __restrict__ A16) {
    int idx = blockIdx.x * blockDim.x + threadIdx.x;      // 0 .. 4M-1
    int h   = idx & 15;
    int L   = (idx >> 4) & 31;
    int kc  = (idx >> 9) & 63;
    int ct  = idx >> 15;
    int k   = kc * 32 + ((L & 16) ? 16 : 0) + h;
    int i   = ct * 16 + (L & 15);
    A16[idx] = (_Float16)A[(size_t)k * NS + i];
}

// alpha_0 stored directly as scaled f16; also reset grid-barrier state.
__global__ void init_alpha(const float* __restrict__ Pi0,
                           const float* __restrict__ Bm,
                           const int*   __restrict__ se,
                           _Float16* __restrict__ a16, int* cnt, int* gen) {
    int i = blockIdx.x * blockDim.x + threadIdx.x;
    if (i < NS) a16[i] = (_Float16)(Pi0[i] * Bm[(size_t)i * NOBS + se[0]] * ALPHA_SCALE);
    if (i == 0) { *cnt = 0; *gen = 0; }
}

// ---------------------------------------------------------------------------
// Persistent forward recurrence.  Block ct owns 16 output columns; its 16
// waves each own 4 K-chunks (register-resident packed-A slab).  Per step:
// issue ALL alpha global_load_b128s (sched_barrier pins them above the
// wmmas), 4-deep wmma_f32_16x16x32_f16 chain, LDS cross-wave reduce,
// emission scale, f16 store, device-wide barrier.
// ---------------------------------------------------------------------------
__global__ void __launch_bounds__(NTHREADS)
hmm_forward(const float* __restrict__ Bm, const int* __restrict__ se,
            const _Float16* __restrict__ A16,
            _Float16* ab0, _Float16* ab1,
            int* cnt, int* gen) {
    __shared__ float partial[NWAVE_TILE][16];

    const int tid  = threadIdx.x;
    const int lane = tid & 31;
    const int wave = tid >> 5;
    const int ct   = blockIdx.x;
    const int hiG  = (lane >= 16) ? 16 : 0;           // A-operand K-group split
    const int kc0  = wave * KC_PER_WAVE;

    union V16 { v16h v; u32x4 q[2]; };

    // register-resident slab of packed A: 4 chunks, contiguous per lane
    V16 btile[KC_PER_WAVE];
    {
        const _Float16* tb = A16 + (((size_t)ct * KC_TOTAL + kc0) * 32 + lane) * 16;
#pragma unroll
        for (int kk = 0; kk < KC_PER_WAVE; ++kk) {
            const u32x4* gp = (const u32x4*)(tb + (size_t)kk * (32 * 16));
            btile[kk].q[0] = gp[0];
            btile[kk].q[1] = gp[1];
        }
    }

    const unsigned long long a0i = (unsigned long long)ab0;
    const unsigned long long a1i = (unsigned long long)ab1;
    const bool redlane = (tid < 16);
    const int  gi      = ct * 16 + (tid & 15);        // output column for redlane

    for (int t = 1; t < TT; ++t) {
        const unsigned long long aci = (t & 1) ? a0i : a1i;  // read buf[(t-1)&1]
        const unsigned long long ani = (t & 1) ? a1i : a0i;  // write buf[t&1]

        // prefetch emission factor for this step's output columns
        float bv = 0.f;
        if (redlane) bv = Bm[(size_t)gi * NOBS + se[t]];

        // issue ALL alpha-chunk loads first; per-lane K layout:
        // halves 0..15 -> kbase+hiG .. kbase+hiG+15
        V16 areg[KC_PER_WAVE];
#pragma unroll
        for (int kk = 0; kk < KC_PER_WAVE; ++kk) {
            const unsigned long long base =
                aci + (unsigned long long)(((kc0 + kk) * 32 + hiG) * 2);
            areg[kk].q[0] = *(g_cu4)(base);
            areg[kk].q[1] = *(g_cu4)(base + 16);
        }
        SCHED_FENCE();   // keep all loads issued before the wmma chain

        // dependent WMMA chain against the register-resident A slab
        v8f acc = {};
#pragma unroll
        for (int kk = 0; kk < KC_PER_WAVE; ++kk) {
            acc = __builtin_amdgcn_wmma_f32_16x16x32_f16(
                false, areg[kk].v, false, btile[kk].v, (short)0, acc, false, false);
        }

        // all 16 rows of D identical; lane L<16 holds column ct*16+L
        if (lane < 16) partial[wave][lane] = acc[0];
        __syncthreads();

        if (redlane) {
            float s = 0.f;
#pragma unroll
            for (int w = 0; w < NWAVE_TILE; ++w) s += partial[w][tid];
            // scale self-propagates: stored value = SCALE * alpha_t
            *(g_h)(ani + (unsigned long long)gi * 2) = (_Float16)(s * bv);
        }

        // ---- device-wide barrier: release, arrive/spin, acquire ----
        __builtin_amdgcn_fence(__ATOMIC_RELEASE, "agent");
        __syncthreads();
        if (tid == 0) {
            int g = __hip_atomic_load(gen, __ATOMIC_RELAXED, __HIP_MEMORY_SCOPE_AGENT);
            int arrived = __hip_atomic_fetch_add(cnt, 1, __ATOMIC_ACQ_REL,
                                                 __HIP_MEMORY_SCOPE_AGENT);
            if (arrived == NBLK - 1) {
                __hip_atomic_store(cnt, 0, __ATOMIC_RELAXED, __HIP_MEMORY_SCOPE_AGENT);
                __hip_atomic_fetch_add(gen, 1, __ATOMIC_RELEASE,
                                       __HIP_MEMORY_SCOPE_AGENT);
            } else {
                while (__hip_atomic_load(gen, __ATOMIC_ACQUIRE,
                                         __HIP_MEMORY_SCOPE_AGENT) == g)
                    __builtin_amdgcn_s_sleep(2);
            }
        }
        __syncthreads();
        __builtin_amdgcn_fence(__ATOMIC_ACQUIRE, "agent");
    }
}

// out[0] = -log( sum_i alpha_T[i] )  (alpha stored as SCALE*alpha in f16)
__global__ void finalize(const _Float16* __restrict__ aF, float* __restrict__ out) {
    __shared__ float red[256];
    const int tid = threadIdx.x;
    float s = 0.f;
    for (int j = tid; j < NS; j += 256) s += (float)aF[j];
    red[tid] = s;
    __syncthreads();
    for (int off = 128; off > 0; off >>= 1) {
        if (tid < off) red[tid] += red[tid + off];
        __syncthreads();
    }
    if (tid == 0) out[0] = -logf(red[0] * ALPHA_INVS);
}

extern "C" void kernel_launch(void* const* d_in, const int* in_sizes, int n_in,
                              void* d_out, int out_size, void* d_ws, size_t ws_size,
                              hipStream_t stream) {
    const float* Pi0 = (const float*)d_in[0];   // [2048]
    const float* A   = (const float*)d_in[1];   // [2048,2048]
    const float* Bm  = (const float*)d_in[2];   // [2048,8192]
    const int*   se  = (const int*)  d_in[3];   // [8192]

    char* ws = (char*)d_ws;
    _Float16* A16  = (_Float16*)ws;                          // 8 MB packed f16 A
    _Float16* a0   = (_Float16*)(ws + (size_t)NS * NS * 2);  // alpha ping (f16)
    _Float16* a1   = a0 + NS;                                // alpha pong (f16)
    int*      cnt  = (int*)(a1 + NS);
    int*      gen  = cnt + 1;

    pack_A16   <<<(NS * NS) / 256, 256, 0, stream>>>(A, A16);
    init_alpha <<<NS / 256,        256, 0, stream>>>(Pi0, Bm, se, a0, cnt, gen);
    hmm_forward<<<NBLK, NTHREADS, 0, stream>>>(Bm, se, A16, a0, a1, cnt, gen);
    // T-1 = 8191 steps -> final alpha in bufs[8191 & 1] = a1
    finalize   <<<1, 256, 0, stream>>>(a1, (float*)d_out);
}
